// LSTMDecoder_7937099563129
// MI455X (gfx1250) — compile-verified, matrix operand order checked by reading, not run on hardware
//
#include <hip/hip_runtime.h>

// Problem sizes (fixed by the reference)
#define H_DIM 1024
#define B_DIM 2048
#define S_DIM 128

typedef __attribute__((ext_vector_type(16))) __bf16 bf16x16;
typedef __attribute__((ext_vector_type(8)))  float  f32x8;

union FragAB {
  bf16x16 v;
  uint4   q[2];
};

__device__ __forceinline__ unsigned short f32_to_bf16(float f) {
  unsigned u = __float_as_uint(f);
  u += 0x7FFFu + ((u >> 16) & 1u);   // round-to-nearest-even
  return (unsigned short)(u >> 16);
}

__device__ __forceinline__ float bf16_to_f32(unsigned short s) {
  return __uint_as_float(((unsigned)s) << 16);
}

__device__ __forceinline__ float sigmoidf_(float x) {
  return 1.0f / (1.0f + __expf(-x));
}

// tanh via a single native exp (v_exp_f32): tanh(x) = (1 - e^-2x)/(1 + e^-2x)
__device__ __forceinline__ float tanhf_(float x) {
  const float e = __expf(-2.0f * x);
  return (1.0f - e) / (1.0f + e);
}

// ---------------------------------------------------------------------------
// Prep: W_hh f32 [4H,H] -> bf16
// ---------------------------------------------------------------------------
__global__ __launch_bounds__(256) void convert_w_kernel(
    const float* __restrict__ W, unsigned short* __restrict__ Wbf, int n) {
  int i = blockIdx.x * 256 + threadIdx.x;
  if (i < n) Wbf[i] = f32_to_bf16(W[i]);
}

// ---------------------------------------------------------------------------
// Prep: h0 -> bf16, c0 -> ws, x -> 0
// ---------------------------------------------------------------------------
__global__ __launch_bounds__(256) void init_state_kernel(
    const float* __restrict__ hidden, const float* __restrict__ cell,
    unsigned short* __restrict__ hbf, float* __restrict__ c,
    float* __restrict__ x, int nBH) {
  int i = blockIdx.x * 256 + threadIdx.x;
  if (i < nBH) {
    hbf[i] = f32_to_bf16(hidden[i]);
    c[i]   = cell[i];
  }
  if (i < B_DIM) x[i] = 0.0f;
}

// ---------------------------------------------------------------------------
// One LSTM step. Block = 256 threads = 8 waves arranged 4(M) x 2(N).
// Each wave computes TWO 16x16 M-subtiles for each of the 4 gates (8 WMMA
// accumulators), sharing the 4 gate B-fragments in registers.
// Block tile = 128(M) x 32(N). Software-pipelined (ping/pong fragment sets).
// ---------------------------------------------------------------------------
__global__ __launch_bounds__(256) void lstm_step_kernel(
    const unsigned short* __restrict__ hbf_in,   // [B,H] bf16
    unsigned short*       __restrict__ hbf_out,  // [B,H] bf16
    float*                __restrict__ c,        // [B,H] f32 (in-place)
    const float*          __restrict__ x,        // [B]    f32
    const unsigned short* __restrict__ Wbf,      // [4H,H] bf16
    const float*          __restrict__ W_ih,     // [4H]
    const float*          __restrict__ b_ih,     // [4H]
    const float*          __restrict__ b_hh)     // [4H]
{
  const int lane = threadIdx.x & 31;
  const int wid  = threadIdx.x >> 5;
  const int m_base = blockIdx.y * 128 + (wid >> 1) * 32;  // 2 subtiles of 16
  const int n_base = blockIdx.x * 32  + (wid & 1) * 16;
  const int half = lane >> 4;   // K group select (0: K0-7/16-23, 1: K8-15/24-31)
  const int r    = lane & 15;   // A: row within tile ; B: col within tile
  const size_t HH = (size_t)H_DIM * H_DIM;   // gate stride in W (elements)

  const unsigned short* ap0 = hbf_in + (size_t)(m_base + r) * H_DIM + (half << 3);
  const unsigned short* ap1 = ap0 + (size_t)16 * H_DIM;
  const unsigned short* bp  = Wbf + (size_t)(n_base + r) * H_DIM + (half << 3);

  f32x8 acc00 = {}, acc01 = {}, acc02 = {}, acc03 = {};   // subtile 0, gates i,f,g,o
  f32x8 acc10 = {}, acc11 = {}, acc12 = {}, acc13 = {};   // subtile 1, gates i,f,g,o

  FragAB aC0, aC1, bC0, bC1, bC2, bC3;   // current set
  FragAB aN0, aN1, bN0, bN1, bN2, bN3;   // next set

#define LOADSET(A0, A1, B0, B1, B2, B3, KB) do {                                   \
    A0.q[0] = *(const uint4*)(ap0 + (KB));          A0.q[1] = *(const uint4*)(ap0 + (KB) + 16); \
    A1.q[0] = *(const uint4*)(ap1 + (KB));          A1.q[1] = *(const uint4*)(ap1 + (KB) + 16); \
    B0.q[0] = *(const uint4*)(bp + 0 * HH + (KB));  B0.q[1] = *(const uint4*)(bp + 0 * HH + (KB) + 16); \
    B1.q[0] = *(const uint4*)(bp + 1 * HH + (KB));  B1.q[1] = *(const uint4*)(bp + 1 * HH + (KB) + 16); \
    B2.q[0] = *(const uint4*)(bp + 2 * HH + (KB));  B2.q[1] = *(const uint4*)(bp + 2 * HH + (KB) + 16); \
    B3.q[0] = *(const uint4*)(bp + 3 * HH + (KB));  B3.q[1] = *(const uint4*)(bp + 3 * HH + (KB) + 16); \
  } while (0)

#define WMMASET(A0, A1, B0, B1, B2, B3) do {                                                          \
    acc00 = __builtin_amdgcn_wmma_f32_16x16x32_bf16(false, A0.v, false, B0.v, (short)0, acc00, false, false); \
    acc10 = __builtin_amdgcn_wmma_f32_16x16x32_bf16(false, A1.v, false, B0.v, (short)0, acc10, false, false); \
    acc01 = __builtin_amdgcn_wmma_f32_16x16x32_bf16(false, A0.v, false, B1.v, (short)0, acc01, false, false); \
    acc11 = __builtin_amdgcn_wmma_f32_16x16x32_bf16(false, A1.v, false, B1.v, (short)0, acc11, false, false); \
    acc02 = __builtin_amdgcn_wmma_f32_16x16x32_bf16(false, A0.v, false, B2.v, (short)0, acc02, false, false); \
    acc12 = __builtin_amdgcn_wmma_f32_16x16x32_bf16(false, A1.v, false, B2.v, (short)0, acc12, false, false); \
    acc03 = __builtin_amdgcn_wmma_f32_16x16x32_bf16(false, A0.v, false, B3.v, (short)0, acc03, false, false); \
    acc13 = __builtin_amdgcn_wmma_f32_16x16x32_bf16(false, A1.v, false, B3.v, (short)0, acc13, false, false); \
  } while (0)

  LOADSET(aC0, aC1, bC0, bC1, bC2, bC3, 0);

#pragma unroll
  for (int kb = 0; kb < H_DIM; kb += 64) {
    LOADSET(aN0, aN1, bN0, bN1, bN2, bN3, kb + 32);     // next stage in flight
    if (kb + 128 < H_DIM) {                             // uniform branch
      __builtin_prefetch(ap0 + kb + 128, 0, 3);
      __builtin_prefetch(ap1 + kb + 128, 0, 3);
      __builtin_prefetch(bp  + kb + 128, 0, 3);
    }
    WMMASET(aC0, aC1, bC0, bC1, bC2, bC3);              // compute current
    if (kb + 64 < H_DIM)
      LOADSET(aC0, aC1, bC0, bC1, bC2, bC3, kb + 64);   // stage after next
    WMMASET(aN0, aN1, bN0, bN1, bN2, bN3);              // compute next
  }
#undef LOADSET
#undef WMMASET

  // Epilogue. C/D layout: lanes 0-15: (M=v, N=lane); lanes 16-31: (M=v+8, N=lane-16).
  const int n = n_base + r;
  const float bi = b_ih[0 * H_DIM + n] + b_hh[0 * H_DIM + n];
  const float bf = b_ih[1 * H_DIM + n] + b_hh[1 * H_DIM + n];
  const float bg = b_ih[2 * H_DIM + n] + b_hh[2 * H_DIM + n];
  const float bo = b_ih[3 * H_DIM + n] + b_hh[3 * H_DIM + n];
  const float wi = W_ih[0 * H_DIM + n];
  const float wf = W_ih[1 * H_DIM + n];
  const float wg = W_ih[2 * H_DIM + n];
  const float wo = W_ih[3 * H_DIM + n];

#define EPILOG(AI, AF, AG, AO, MOFF) do {                                   \
    _Pragma("unroll")                                                       \
    for (int v = 0; v < 8; ++v) {                                           \
      const int m = m_base + (MOFF) + (half << 3) + v;                      \
      const float xv = x[m];                                                \
      const float ig = sigmoidf_(AI[v] + xv * wi + bi);                     \
      const float fg = sigmoidf_(AF[v] + xv * wf + bf);                     \
      const float gg = tanhf_   (AG[v] + xv * wg + bg);                     \
      const float og = sigmoidf_(AO[v] + xv * wo + bo);                     \
      const size_t idx = (size_t)m * H_DIM + n;                             \
      const float cn = fg * c[idx] + ig * gg;                               \
      c[idx] = cn;                                                          \
      hbf_out[idx] = f32_to_bf16(og * tanhf_(cn));                          \
    }                                                                       \
  } while (0)

  EPILOG(acc00, acc01, acc02, acc03, 0);
  EPILOG(acc10, acc11, acc12, acc13, 16);
#undef EPILOG
}

// ---------------------------------------------------------------------------
// Per-step head: pred[b] = h[b,:] . W_fc[t] + b_fc[t]; feeds back as next x.
// One block per batch row; block reduction in LDS.
// ---------------------------------------------------------------------------
__global__ __launch_bounds__(256) void lstm_pred_kernel(
    const unsigned short* __restrict__ hbf,   // [B,H] bf16 (h_new)
    const float* __restrict__ W_fc,           // [S,H]
    const float* __restrict__ b_fc,           // [S]
    float* __restrict__ out,                  // [B,S]
    float* __restrict__ x,                    // [B]
    int t)
{
  const int b   = blockIdx.x;
  const int tid = threadIdx.x;
  const unsigned short* hr = hbf + (size_t)b * H_DIM;
  const float* w = W_fc + (size_t)t * H_DIM;

  float s = 0.0f;
#pragma unroll
  for (int h = tid; h < H_DIM; h += 256)
    s += bf16_to_f32(hr[h]) * w[h];

  __shared__ float sm[256];
  sm[tid] = s;
  __syncthreads();
#pragma unroll
  for (int off = 128; off > 0; off >>= 1) {
    if (tid < off) sm[tid] += sm[tid + off];
    __syncthreads();
  }
  if (tid == 0) {
    const float p = sm[0] + b_fc[t];
    out[(size_t)b * S_DIM + t] = p;
    x[b] = p;
  }
}

// ---------------------------------------------------------------------------
extern "C" void kernel_launch(void* const* d_in, const int* in_sizes, int n_in,
                              void* d_out, int out_size, void* d_ws, size_t ws_size,
                              hipStream_t stream) {
  (void)in_sizes; (void)n_in; (void)out_size; (void)ws_size;
  const float* hidden = (const float*)d_in[0];   // [1,B,H]
  const float* cell   = (const float*)d_in[1];   // [1,B,H]
  const float* W_ih   = (const float*)d_in[2];   // [4H,1]
  const float* W_hh   = (const float*)d_in[3];   // [4H,H]
  const float* b_ih   = (const float*)d_in[4];   // [4H]
  const float* b_hh   = (const float*)d_in[5];   // [4H]
  const float* W_fc   = (const float*)d_in[6];   // [S,H]
  const float* b_fc   = (const float*)d_in[7];   // [S]
  float* out = (float*)d_out;                    // [B,S]

  // Workspace layout (bytes):
  //   [0,8M)      W_hh bf16  (4H*H*2 = 8 MB)
  //   [8M,12M)    hbf0       (B*H*2  = 4 MB)
  //   [12M,16M)   hbf1       (B*H*2  = 4 MB)
  //   [16M,24M)   c f32      (B*H*4  = 8 MB)
  //   [24M,..)    x f32      (B*4    = 8 KB)
  char* ws = (char*)d_ws;
  unsigned short* Wbf  = (unsigned short*)(ws);
  unsigned short* hbf0 = (unsigned short*)(ws + ((size_t)8  << 20));
  unsigned short* hbf1 = (unsigned short*)(ws + ((size_t)12 << 20));
  float*          cbuf = (float*)(ws + ((size_t)16 << 20));
  float*          xbuf = (float*)(ws + ((size_t)24 << 20));

  const int nW  = 4 * H_DIM * H_DIM;
  const int nBH = B_DIM * H_DIM;
  convert_w_kernel<<<(nW + 255) / 256, 256, 0, stream>>>(W_hh, Wbf, nW);
  init_state_kernel<<<(nBH + 255) / 256, 256, 0, stream>>>(hidden, cell, hbf0, cbuf, xbuf, nBH);

  dim3 gridA(H_DIM / 32, B_DIM / 128);  // 32 x 16 workgroups
  unsigned short* hin  = hbf0;
  unsigned short* hout = hbf1;
  for (int t = 0; t < S_DIM; ++t) {
    lstm_step_kernel<<<gridA, 256, 0, stream>>>(hin, hout, cbuf, xbuf, Wbf, W_ih, b_ih, b_hh);
    lstm_pred_kernel<<<B_DIM, 256, 0, stream>>>(hout, W_fc, b_fc, out, xbuf, t);
    unsigned short* tmp = hin; hin = hout; hout = tmp;
  }
}
